// IsaacVisionTransformer_63599875719455
// MI455X (gfx1250) — compile-verified, matrix operand order checked by reading, not run on hardware
//
#include <hip/hip_runtime.h>
#include <hip/hip_bf16.h>
#include <math.h>

// ---------------- static geometry ----------------
#define L_TOK   2048
#define HID     768
#define HEADS   12
#define HD      64
#define FF      3072
#define QKV_N   (3 * HID)      // 2304
#define SMAX    1024           // widest attention block
#define POSN    16
#define NL      2
#define OUT_TOK 512            // L / 4

typedef __bf16 bf16_t;
typedef __attribute__((ext_vector_type(16))) __bf16 v16bf;
typedef __attribute__((ext_vector_type(8)))  __bf16 v8bf;
typedef __attribute__((ext_vector_type(8)))  float  v8f;

__constant__ int c_img_off[4] = {0, 1024, 1536, 1792};
__constant__ int c_img_h[4]   = {32, 16, 16, 16};
__constant__ int c_img_w[4]   = {32, 32, 16, 16};
__constant__ int c_mrg_off[4] = {0, 256, 384, 448};

// ---------------- helpers ----------------
__device__ __forceinline__ float gelu_tanh(float x) {
    float x3 = x * x * x;
    return 0.5f * x * (1.0f + tanhf(0.7978845608028654f * (x + 0.044715f * x3)));
}

// Load one 16-bf16 WMMA fragment: two contiguous 16-byte chunks
// (K = koff..koff+7 and K = koff+16..koff+23 relative to tile start).
__device__ __forceinline__ v16bf load_frag16(const bf16_t* p) {
    v8bf lo = *reinterpret_cast<const v8bf*>(p);
    v8bf hi = *reinterpret_cast<const v8bf*>(p + 16);
    v16bf r;
#pragma unroll
    for (int i = 0; i < 8; ++i) { r[i] = lo[i]; r[i + 8] = hi[i]; }
    return r;
}

// ---------------- conversion kernels ----------------
__global__ void k_f32_to_bf16(const float* __restrict__ x, bf16_t* __restrict__ y, int n) {
    int i = blockIdx.x * blockDim.x + threadIdx.x;
    if (i < n) y[i] = (bf16_t)x[i];
}

// W: K x N (row-major, fp32) -> Wt: N x K (row-major, bf16)
__global__ void k_transpose_f32_to_bf16(const float* __restrict__ W, bf16_t* __restrict__ Wt,
                                        int K, int N) {
    int n = blockIdx.x * 16 + threadIdx.x;
    int k = blockIdx.y * 16 + threadIdx.y;
    if (n < N && k < K) Wt[(size_t)n * K + k] = (bf16_t)W[(size_t)k * N + n];
}

// V slice of the fused QKV activation (L x 2304, bf16) -> Vt (768 x L, bf16)
__global__ void k_v_transpose(const bf16_t* __restrict__ qkvb, bf16_t* __restrict__ vt) {
    int l = blockIdx.x * 16 + threadIdx.x;
    int d = blockIdx.y * 16 + threadIdx.y;
    if (l < L_TOK && d < HID)
        vt[(size_t)d * L_TOK + l] = qkvb[(size_t)l * QKV_N + 2 * HID + d];
}

// ---------------- bf16 WMMA GEMM ----------------
// C[M,N] = act( alpha * A[M,K] @ Bt[N,K]^T + bias + residual )
// One wave -> 32x64 output tile: 2 A fragments x 4 B fragments = 8 WMMAs
// per K-step (8 WMMA per 12 b128 loads). Requires M%32==0, N%64==0, K%32==0.
__global__ __launch_bounds__(128)
void k_gemm_bf16(const bf16_t* __restrict__ A, const bf16_t* __restrict__ Bt,
                 const float* __restrict__ bias, const float* __restrict__ residual,
                 float* __restrict__ Cf, bf16_t* __restrict__ Cb,
                 int M, int N, int K, int lda, int ldb, int ldc,
                 long long strideA, long long strideB, long long strideC,
                 float alpha, int act) {
    const int batch = blockIdx.z;
    A  += (size_t)batch * strideA;
    Bt += (size_t)batch * strideB;
    const size_t coff = (size_t)batch * strideC;

    const int wave = blockIdx.x * (blockDim.x >> 5) + (threadIdx.x >> 5);
    const int lane = threadIdx.x & 31;
    const int ntile4 = N >> 6;
    const int total  = (M >> 5) * ntile4;
    if (wave >= total) return;                    // wave-uniform: EXEC stays all-ones
    const int tm  = wave / ntile4;
    const int tn4 = wave % ntile4;

    const int arow0 = tm * 32 + (lane & 15);      // m-tile 0 row for this lane
    const int koff  = (lane < 16) ? 0 : 8;        // ISA 7.12.2 16-bit A/B lane layout

    v8f acc[2][4];
#pragma unroll
    for (int mi = 0; mi < 2; ++mi)
#pragma unroll
        for (int t = 0; t < 4; ++t)
#pragma unroll
            for (int r = 0; r < 8; ++r) acc[mi][t][r] = 0.0f;

    for (int k0 = 0; k0 < K; k0 += 32) {
        v16bf af0 = load_frag16(A + (size_t)arow0 * lda + k0 + koff);
        v16bf af1 = load_frag16(A + (size_t)(arow0 + 16) * lda + k0 + koff);
#pragma unroll
        for (int t = 0; t < 4; ++t) {
            int brow = tn4 * 64 + t * 16 + (lane & 15);
            v16bf bfr = load_frag16(Bt + (size_t)brow * ldb + k0 + koff);
            acc[0][t] = __builtin_amdgcn_wmma_f32_16x16x32_bf16(
                false, af0, false, bfr, (short)0, acc[0][t], false, false);
            acc[1][t] = __builtin_amdgcn_wmma_f32_16x16x32_bf16(
                false, af1, false, bfr, (short)0, acc[1][t], false, false);
        }
    }

#pragma unroll
    for (int mi = 0; mi < 2; ++mi) {
#pragma unroll
        for (int t = 0; t < 4; ++t) {
            int col = tn4 * 64 + t * 16 + (lane & 15);
            float bc = bias ? bias[col] : 0.0f;
#pragma unroll
            for (int r = 0; r < 8; ++r) {
                int row = tm * 32 + mi * 16 + ((lane < 16) ? r : r + 8);
                float v = acc[mi][t][r] * alpha + bc;
                size_t o = coff + (size_t)row * ldc + col;
                if (residual) v += residual[o];
                if (act == 1) v = gelu_tanh(v);
                if (Cf) Cf[o] = v;
                if (Cb) Cb[o] = (bf16_t)v;
            }
        }
    }
}

// ---------------- layernorm (wave per row) ----------------
__global__ void k_layernorm(const float* __restrict__ x, const float* __restrict__ w,
                            const float* __restrict__ b, float* __restrict__ yf,
                            bf16_t* __restrict__ yb) {
    int row  = blockIdx.x * (blockDim.x >> 5) + (threadIdx.x >> 5);
    int lane = threadIdx.x & 31;
    if (row >= L_TOK) return;
    const float* xr = x + (size_t)row * HID;
    float s = 0.f, s2 = 0.f;
    for (int i = lane; i < HID; i += 32) { float v = xr[i]; s += v; s2 += v * v; }
    for (int m = 16; m > 0; m >>= 1) { s += __shfl_xor(s, m, 32); s2 += __shfl_xor(s2, m, 32); }
    float mean = s * (1.0f / HID);
    float var  = s2 * (1.0f / HID) - mean * mean;
    float rstd = rsqrtf(var + 1e-6f);
    for (int i = lane; i < HID; i += 32) {
        float v = (xr[i] - mean) * rstd * w[i] + b[i];
        if (yf) yf[(size_t)row * HID + i] = v;
        if (yb) yb[(size_t)row * HID + i] = (bf16_t)v;
    }
}

// ---------------- positional embedding add (bilinear resize, static grid) ----------------
__global__ void k_pos_add(float* __restrict__ x, const float* __restrict__ pos_emb) {
    int t = blockIdx.x;
    int img = (t >= 1792) ? 3 : (t >= 1536) ? 2 : (t >= 1024) ? 1 : 0;
    int off = c_img_off[img], h = c_img_h[img], w = c_img_w[img];
    int r = (t - off) / w, c = (t - off) % w;

    float fy = (r + 0.5f) * ((float)POSN / h) - 0.5f;
    int jy = (int)floorf(fy); float ay = fy - (float)jy;
    int y0, y1; float wy0, wy1;
    if (jy < 0)             { y0 = y1 = 0;        wy0 = 1.f; wy1 = 0.f; }
    else if (jy >= POSN - 1){ y0 = y1 = POSN - 1; wy0 = 1.f; wy1 = 0.f; }
    else                    { y0 = jy; y1 = jy + 1; wy0 = 1.f - ay; wy1 = ay; }

    float fx = (c + 0.5f) * ((float)POSN / w) - 0.5f;
    int jx = (int)floorf(fx); float ax = fx - (float)jx;
    int x0, x1; float wx0, wx1;
    if (jx < 0)             { x0 = x1 = 0;        wx0 = 1.f; wx1 = 0.f; }
    else if (jx >= POSN - 1){ x0 = x1 = POSN - 1; wx0 = 1.f; wx1 = 0.f; }
    else                    { x0 = jx; x1 = jx + 1; wx0 = 1.f - ax; wx1 = ax; }

    const float* g00 = pos_emb + (size_t)(y0 * POSN + x0) * HID;
    const float* g01 = pos_emb + (size_t)(y0 * POSN + x1) * HID;
    const float* g10 = pos_emb + (size_t)(y1 * POSN + x0) * HID;
    const float* g11 = pos_emb + (size_t)(y1 * POSN + x1) * HID;
    for (int ch = threadIdx.x; ch < HID; ch += blockDim.x) {
        float p = wy0 * (wx0 * g00[ch] + wx1 * g01[ch]) +
                  wy1 * (wx0 * g10[ch] + wx1 * g11[ch]);
        x[(size_t)t * HID + ch] += p;
    }
}

// ---------------- block softmax (wave per (head, token) row, in place) ----------------
__global__ void k_softmax_block(bf16_t* __restrict__ scores) {
    int idx  = blockIdx.x * (blockDim.x >> 5) + (threadIdx.x >> 5);
    int lane = threadIdx.x & 31;
    if (idx >= HEADS * L_TOK) return;
    int t = idx % L_TOK;
    int img = (t >= 1792) ? 3 : (t >= 1536) ? 2 : (t >= 1024) ? 1 : 0;
    int S = (img == 0) ? 1024 : (img == 1) ? 512 : 256;
    bf16_t* row = scores + (size_t)idx * SMAX;

    float mx = -3.0e38f;
    for (int j = lane; j < S; j += 32) mx = fmaxf(mx, (float)row[j]);
    for (int m = 16; m > 0; m >>= 1) mx = fmaxf(mx, __shfl_xor(mx, m, 32));
    float sum = 0.f;
    for (int j = lane; j < S; j += 32) sum += __expf((float)row[j] - mx);
    for (int m = 16; m > 0; m >>= 1) sum += __shfl_xor(sum, m, 32);
    float inv = 1.0f / sum;
    for (int j = lane; j < S; j += 32)
        row[j] = (bf16_t)(__expf((float)row[j] - mx) * inv);
}

// ---------------- pixel-shuffle varlen merge ----------------
__global__ void k_pixel_shuffle(const float* __restrict__ y, float* __restrict__ out) {
    int t = blockIdx.x;                    // merged token 0..511
    int img = (t >= 448) ? 3 : (t >= 384) ? 2 : (t >= 256) ? 1 : 0;
    int m = t - c_mrg_off[img];
    int off = c_img_off[img], w = c_img_w[img];
    int gw = w >> 1;
    int gr = m / gw, gc = m % gw;
    for (int ch = threadIdx.x; ch < HID; ch += blockDim.x) {
#pragma unroll
        for (int g = 0; g < 4; ++g) {
            int dr = g >> 1, dc = g & 1;
            int src = off + (2 * gr + dr) * w + (2 * gc + dc);
            out[(size_t)t * (HID * 4) + g * HID + ch] = y[(size_t)src * HID + ch];
        }
    }
}

// ---------------- host orchestration ----------------
extern "C" void kernel_launch(void* const* d_in, const int* in_sizes, int n_in,
                              void* d_out, int out_size, void* d_ws, size_t ws_size,
                              hipStream_t stream) {
    (void)in_sizes; (void)n_in; (void)out_size; (void)ws_size;

    const float* seq_patches = (const float*)d_in[0];
    const float* patch_w     = (const float*)d_in[2];
    const float* patch_b     = (const float*)d_in[3];
    const float* pos_emb     = (const float*)d_in[4];
    const float* ln1_w       = (const float*)d_in[5];
    const float* ln1_b       = (const float*)d_in[6];
    const float* q_w         = (const float*)d_in[7];
    const float* q_b         = (const float*)d_in[8];
    const float* k_w         = (const float*)d_in[9];
    const float* k_b         = (const float*)d_in[10];
    const float* v_w         = (const float*)d_in[11];
    const float* v_b         = (const float*)d_in[12];
    const float* o_w         = (const float*)d_in[13];
    const float* o_b         = (const float*)d_in[14];
    const float* ln2_w       = (const float*)d_in[15];
    const float* ln2_b       = (const float*)d_in[16];
    const float* fc1_w       = (const float*)d_in[17];
    const float* fc1_b       = (const float*)d_in[18];
    const float* fc2_w       = (const float*)d_in[19];
    const float* fc2_b       = (const float*)d_in[20];
    const float* postln_w    = (const float*)d_in[21];
    const float* postln_b    = (const float*)d_in[22];
    float* out = (float*)d_out;

    // ---- workspace carve ----
    char* wp = (char*)d_ws;
    auto alloc = [&](size_t bytes) -> void* {
        void* r = (void*)wp;
        wp += (bytes + 255) & ~(size_t)255;
        return r;
    };
    bf16_t* seqb   = (bf16_t*)alloc((size_t)L_TOK * HID * 2);
    bf16_t* patchT = (bf16_t*)alloc((size_t)HID * HID * 2);
    bf16_t* qT[NL], *kT[NL], *vT[NL], *oT[NL], *f1T[NL], *f2T[NL];
    for (int i = 0; i < NL; ++i) {
        qT[i]  = (bf16_t*)alloc((size_t)HID * HID * 2);
        kT[i]  = (bf16_t*)alloc((size_t)HID * HID * 2);
        vT[i]  = (bf16_t*)alloc((size_t)HID * HID * 2);
        oT[i]  = (bf16_t*)alloc((size_t)HID * HID * 2);
        f1T[i] = (bf16_t*)alloc((size_t)HID * FF * 2);
        f2T[i] = (bf16_t*)alloc((size_t)FF * HID * 2);
    }
    float*  xf     = (float*)alloc((size_t)L_TOK * HID * 4);
    bf16_t* lnb    = (bf16_t*)alloc((size_t)L_TOK * HID * 2);
    bf16_t* qkvb   = (bf16_t*)alloc((size_t)L_TOK * QKV_N * 2);
    bf16_t* vtb    = (bf16_t*)alloc((size_t)HID * L_TOK * 2);
    bf16_t* scores = (bf16_t*)alloc((size_t)HEADS * L_TOK * SMAX * 2);
    bf16_t* attnb  = (bf16_t*)alloc((size_t)L_TOK * HID * 2);
    bf16_t* h1b    = (bf16_t*)alloc((size_t)L_TOK * FF * 2);
    float*  yf     = (float*)alloc((size_t)L_TOK * HID * 4);

    auto gemm = [&](const bf16_t* A, const bf16_t* Bt, const float* bias, const float* res,
                    float* Cf, bf16_t* Cb, int M, int N, int K, int lda, int ldb, int ldc,
                    long long sA, long long sB, long long sC, int nb, float alpha, int act) {
        int waves = (M / 32) * (N / 64);
        dim3 g((waves + 3) / 4, 1, nb);
        k_gemm_bf16<<<g, 128, 0, stream>>>(A, Bt, bias, res, Cf, Cb, M, N, K,
                                           lda, ldb, ldc, sA, sB, sC, alpha, act);
    };
    auto transpose = [&](const float* W, bf16_t* Wt, int K, int N) {
        k_transpose_f32_to_bf16<<<dim3(N / 16, K / 16), dim3(16, 16), 0, stream>>>(W, Wt, K, N);
    };

    // ---- weight conversion (fp32 -> transposed bf16, N-major for WMMA B) ----
    transpose(patch_w, patchT, HID, HID);
    for (int i = 0; i < NL; ++i) {
        transpose(q_w  + (size_t)i * HID * HID, qT[i],  HID, HID);
        transpose(k_w  + (size_t)i * HID * HID, kT[i],  HID, HID);
        transpose(v_w  + (size_t)i * HID * HID, vT[i],  HID, HID);
        transpose(o_w  + (size_t)i * HID * HID, oT[i],  HID, HID);
        transpose(fc1_w + (size_t)i * HID * FF, f1T[i], HID, FF);
        transpose(fc2_w + (size_t)i * FF * HID, f2T[i], FF, HID);
    }
    {
        int n = L_TOK * HID;
        k_f32_to_bf16<<<(n + 255) / 256, 256, 0, stream>>>(seq_patches, seqb, n);
    }

    // ---- embeddings: x = seq @ patch_w + patch_b + pos ----
    gemm(seqb, patchT, patch_b, nullptr, xf, nullptr,
         L_TOK, HID, HID, HID, HID, HID, 0, 0, 0, 1, 1.0f, 0);
    k_pos_add<<<L_TOK, 256, 0, stream>>>(xf, pos_emb);

    const int boff[4] = {0, 1024, 1536, 1792};
    const int bsz[4]  = {1024, 512, 256, 256};
    const float scaling = 0.125f;   // 64^-0.5

    for (int i = 0; i < NL; ++i) {
        // LN1 -> bf16
        k_layernorm<<<L_TOK / 4, 128, 0, stream>>>(xf, ln1_w + i * HID, ln1_b + i * HID,
                                                   nullptr, lnb);
        // fused QKV activation buffer (L x 2304), three GEMMs with column offsets
        gemm(lnb, qT[i], q_b + i * HID, nullptr, nullptr, qkvb + 0,
             L_TOK, HID, HID, HID, HID, QKV_N, 0, 0, 0, 1, 1.0f, 0);
        gemm(lnb, kT[i], k_b + i * HID, nullptr, nullptr, qkvb + HID,
             L_TOK, HID, HID, HID, HID, QKV_N, 0, 0, 0, 1, 1.0f, 0);
        gemm(lnb, vT[i], v_b + i * HID, nullptr, nullptr, qkvb + 2 * HID,
             L_TOK, HID, HID, HID, HID, QKV_N, 0, 0, 0, 1, 1.0f, 0);
        // V^T (HEADS*HD x L) for P@V
        k_v_transpose<<<dim3(L_TOK / 16, HID / 16), dim3(16, 16), 0, stream>>>(qkvb, vtb);

        // scores = scale * Q @ K^T, per block, batched over heads
        for (int b = 0; b < 4; ++b) {
            const bf16_t* Aq = qkvb + (size_t)boff[b] * QKV_N;            // + h*HD via strideA
            const bf16_t* Bk = qkvb + (size_t)boff[b] * QKV_N + HID;      // + h*HD via strideB
            bf16_t* Cs = scores + (size_t)boff[b] * SMAX;                 // + h*L*SMAX via strideC
            gemm(Aq, Bk, nullptr, nullptr, nullptr, Cs,
                 bsz[b], bsz[b], HD, QKV_N, QKV_N, SMAX,
                 HD, HD, (long long)L_TOK * SMAX, HEADS, scaling, 0);
        }
        // softmax within each block row
        k_softmax_block<<<(HEADS * L_TOK) / 4, 128, 0, stream>>>(scores);
        // attn = P @ V, per block, batched over heads
        for (int b = 0; b < 4; ++b) {
            const bf16_t* Ap = scores + (size_t)boff[b] * SMAX;
            const bf16_t* Bv = vtb + boff[b];                             // + h*HD*L via strideB
            bf16_t* Ca = attnb + (size_t)boff[b] * HID;                   // + h*HD via strideC
            gemm(Ap, Bv, nullptr, nullptr, nullptr, Ca,
                 bsz[b], HD, bsz[b], SMAX, L_TOK, HID,
                 (long long)L_TOK * SMAX, (long long)HD * L_TOK, HD, HEADS, 1.0f, 0);
        }
        // x = x + attn @ o_w + o_b   (residual in fp32, in place)
        gemm(attnb, oT[i], o_b + i * HID, xf, xf, nullptr,
             L_TOK, HID, HID, HID, HID, HID, 0, 0, 0, 1, 1.0f, 0);
        // MLP
        k_layernorm<<<L_TOK / 4, 128, 0, stream>>>(xf, ln2_w + i * HID, ln2_b + i * HID,
                                                   nullptr, lnb);
        gemm(lnb, f1T[i], fc1_b + i * FF, nullptr, nullptr, h1b,
             L_TOK, FF, HID, HID, HID, FF, 0, 0, 0, 1, 1.0f, 1 /*gelu*/);
        gemm(h1b, f2T[i], fc2_b + i * HID, xf, xf, nullptr,
             L_TOK, HID, FF, FF, FF, HID, 0, 0, 0, 1, 1.0f, 0);
    }

    // post-LN (fp32 out) + pixel-shuffle merge
    k_layernorm<<<L_TOK / 4, 128, 0, stream>>>(xf, postln_w, postln_b, yf, nullptr);
    k_pixel_shuffle<<<OUT_TOK, 256, 0, stream>>>(yf, out);
}
